// MoE_180388626878
// MI455X (gfx1250) — matrix operand order repaired and checked by reading, not compiled, not run on hardware
//
#include <hip/hip_runtime.h>
#include <hip/hip_bf16.h>

// ---------------- problem constants ----------------
#define D_MODEL  1024
#define N_EXP    8
#define T_TOK    8192          // B*S = 4*2048
#define MT       64            // token rows per GEMM block tile (4 subtiles of 16)
#define NT       128           // output cols per block (8 waves x 16)
#define KT       32            // K step per WMMA
#define NSTEP    (D_MODEL / KT)
#define APITCH   40            // shorts: 80B rows -> 16B-aligned frag reads
#define BPITCH   36            // shorts: 72B rows -> 8B-aligned b64 stores

typedef __attribute__((ext_vector_type(16))) __bf16 v16bf;
typedef __attribute__((ext_vector_type(8)))  float  v8f;

union Frag16 { v16bf v; unsigned short u[16]; };

// pack two f32 -> two bf16 (round-half-up + v_perm_b32): 3 VALU total
static __device__ __forceinline__ unsigned int pk2(float lo, float hi) {
    const unsigned int u0 = __float_as_uint(lo) + 0x8000u;
    const unsigned int u1 = __float_as_uint(hi) + 0x8000u;
    // result bytes (LSB..MSB) = {u0.b2, u0.b3, u1.b2, u1.b3}
    return __builtin_amdgcn_perm(u1, u0, 0x07060302u);
}

// ---------------- kernel 1: gating (one wave32 per token) ----------------
__global__ __launch_bounds__(256) void moe_gate_kernel(
    const float* __restrict__ x, const float* __restrict__ gate_w,
    int* __restrict__ topk_e, float* __restrict__ topk_w)
{
    const int lane = threadIdx.x & 31;
    const int tok  = blockIdx.x * 8 + (threadIdx.x >> 5);

    const float4* xr4 = (const float4*)(x + (size_t)tok * D_MODEL);
    float acc[N_EXP];
#pragma unroll
    for (int e = 0; e < N_EXP; ++e) acc[e] = 0.f;

#pragma unroll
    for (int i = 0; i < D_MODEL / 128; ++i) {       // 8 iters, float4 per lane
        const float4 xv = xr4[i * 32 + lane];
#pragma unroll
        for (int e = 0; e < N_EXP; ++e) {
            const float4 gv = ((const float4*)(gate_w + e * D_MODEL))[i * 32 + lane];
            acc[e] = fmaf(xv.x, gv.x, fmaf(xv.y, gv.y, fmaf(xv.z, gv.z, fmaf(xv.w, gv.w, acc[e]))));
        }
    }
#pragma unroll
    for (int e = 0; e < N_EXP; ++e) {
#pragma unroll
        for (int off = 16; off > 0; off >>= 1)
            acc[e] += __shfl_xor(acc[e], off, 32);
    }

    if (lane == 0) {
        int e0 = 0; float v0 = acc[0];
#pragma unroll
        for (int e = 1; e < N_EXP; ++e)            // strict > : first max wins (jax top_k)
            if (acc[e] > v0) { v0 = acc[e]; e0 = e; }
        int e1 = (e0 == 0) ? 1 : 0; float v1 = acc[e1];
#pragma unroll
        for (int e = 0; e < N_EXP; ++e)
            if (e != e0 && acc[e] > v1) { v1 = acc[e]; e1 = e; }
        const float z = __expf(v1 - v0);
        const float s = 1.f / (1.f + z);
        topk_e[tok * 2 + 0] = e0;  topk_e[tok * 2 + 1] = e1;
        topk_w[tok * 2 + 0] = s;   topk_w[tok * 2 + 1] = z * s;
    }
}

// ------- kernel 2: deterministic per-expert compaction (1 wave / expert) -------
__global__ __launch_bounds__(32) void moe_compact_kernel(
    const int* __restrict__ topk_e, const float* __restrict__ topk_w,
    int* __restrict__ counts, int* __restrict__ tok_list, float* __restrict__ w_list)
{
    const int e    = blockIdx.x;
    const int lane = threadIdx.x;
    int offset = 0;
    for (int base = 0; base < T_TOK; base += 32) {
        const int t  = base + lane;
        const bool fa = (topk_e[t * 2 + 0] == e);
        const bool fb = (topk_e[t * 2 + 1] == e);
        const bool f  = fa || fb;
        const float w = fa ? topk_w[t * 2 + 0] : topk_w[t * 2 + 1];
        const unsigned int m32 = (unsigned int)__ballot(f);
        const int pre = __popc(m32 & ((1u << lane) - 1u));
        if (f) {
            tok_list[e * T_TOK + offset + pre] = t;
            w_list [e * T_TOK + offset + pre] = w;
        }
        offset += __popc(m32);
    }
    if (lane == 0) counts[e] = offset;
}

// ------- kernel 3: grouped expert GEMM, bf16 WMMA, ping-pong pipelined staging -------
__global__ __launch_bounds__(256) void moe_expert_gemm_kernel(
    const float* __restrict__ x, const float* __restrict__ weight,
    const int* __restrict__ counts, const int* __restrict__ tok_list,
    const float* __restrict__ w_list, float* __restrict__ out)
{
    const int e     = blockIdx.z;
    const int rbase = blockIdx.y * MT;
    const int nbase = blockIdx.x * NT;
    const int cnt   = counts[e];                 // uniform
    if (rbase >= cnt) return;

    __shared__ unsigned short ldsA[2][MT * APITCH]; // A tile bf16 [row][k], ping-pong
    __shared__ unsigned short ldsB[2][NT * BPITCH]; // B tile bf16 transposed [n][k]
    __shared__ int   stok[MT];
    __shared__ float sw[MT];

    const int tid  = threadIdx.x;
    const int lane = tid & 31;
    const int wv   = tid >> 5;

    if (tid < MT) {
        const int slot = rbase + tid;
        const bool v   = slot < cnt;
        stok[tid] = v ? tok_list[e * T_TOK + slot] : 0;
        sw[tid]   = v ? w_list [e * T_TOK + slot] : 0.f;
    }
    __syncthreads();

    const float* We = weight + (size_t)e * D_MODEL * D_MODEL;  // [K][N]

    // ---- per-thread staging assignments (hoisted pointers) ----
    const int r0 = tid >> 3, r1 = r0 + 32, c4 = (tid & 7) * 4;
    const float4* pA0 = (const float4*)(x + (size_t)stok[r0] * D_MODEL + c4);
    const float4* pA1 = (const float4*)(x + (size_t)stok[r1] * D_MODEL + c4);
    const int kg = (tid >> 5) * 4, n4 = (tid & 31) * 4;
    const float* pB = We + (size_t)kg * D_MODEL + nbase + n4;

    // ---- fragment read indices (constant across K loop) ----
    const int rl   = lane & 15;
    const int hk   = (lane >> 4) * 8;            // A: K-half select per ISA layout
    const int ncol = wv * 16 + rl;
    const int kb   = (lane >> 4) * 8;            // B: K-half select

    v8f c0 = {}, c1 = {}, c2 = {}, c3 = {};

    // prefetch registers
    float4 av0 = pA0[0], av1 = pA1[0];
    float4 bv0 = *(const float4*)(pB + 0 * D_MODEL);
    float4 bv1 = *(const float4*)(pB + 1 * D_MODEL);
    float4 bv2 = *(const float4*)(pB + 2 * D_MODEL);
    float4 bv3 = *(const float4*)(pB + 3 * D_MODEL);

    // one stage+compute body on buffer `buf` for K-step `step`
    auto body = [&](int step, int buf) {
        uint2* sA0 = (uint2*)&ldsA[buf][r0 * APITCH + c4];
        uint2* sA1 = (uint2*)&ldsA[buf][r1 * APITCH + c4];
        uint2* sB0 = (uint2*)&ldsB[buf][(n4 + 0) * BPITCH + kg];
        uint2* sB1 = (uint2*)&ldsB[buf][(n4 + 1) * BPITCH + kg];
        uint2* sB2 = (uint2*)&ldsB[buf][(n4 + 2) * BPITCH + kg];
        uint2* sB3 = (uint2*)&ldsB[buf][(n4 + 3) * BPITCH + kg];

        uint2 t;
        t.x = pk2(av0.x, av0.y); t.y = pk2(av0.z, av0.w); *sA0 = t;
        t.x = pk2(av1.x, av1.y); t.y = pk2(av1.z, av1.w); *sA1 = t;
        t.x = pk2(bv0.x, bv1.x); t.y = pk2(bv2.x, bv3.x); *sB0 = t;
        t.x = pk2(bv0.y, bv1.y); t.y = pk2(bv2.y, bv3.y); *sB1 = t;
        t.x = pk2(bv0.z, bv1.z); t.y = pk2(bv2.z, bv3.z); *sB2 = t;
        t.x = pk2(bv0.w, bv1.w); t.y = pk2(bv2.w, bv3.w); *sB3 = t;
        __syncthreads();   // single barrier per step (ping-pong makes WAR safe)

        if (step + 1 < NSTEP) {                  // prefetch next K-step
            const int koff = (step + 1) * KT;
            av0 = pA0[koff / 4];
            av1 = pA1[koff / 4];
            const float* pBk = pB + (size_t)koff * D_MODEL;
            bv0 = *(const float4*)(pBk + 0 * D_MODEL);
            bv1 = *(const float4*)(pBk + 1 * D_MODEL);
            bv2 = *(const float4*)(pBk + 2 * D_MODEL);
            bv3 = *(const float4*)(pBk + 3 * D_MODEL);
        }

        const unsigned short* fA = &ldsA[buf][rl * APITCH + hk];
        const unsigned short* fB = &ldsB[buf][ncol * BPITCH + kb];
        Frag16 a0, a1, a2, a3, b;
#pragma unroll
        for (int j = 0; j < 8; ++j) {
            a0.u[j] = fA[ 0 * APITCH + j];      a0.u[8 + j] = fA[ 0 * APITCH + 16 + j];
            a1.u[j] = fA[16 * APITCH + j];      a1.u[8 + j] = fA[16 * APITCH + 16 + j];
            a2.u[j] = fA[32 * APITCH + j];      a2.u[8 + j] = fA[32 * APITCH + 16 + j];
            a3.u[j] = fA[48 * APITCH + j];      a3.u[8 + j] = fA[48 * APITCH + 16 + j];
        }
#pragma unroll
        for (int j = 0; j < 8; ++j) { b.u[j] = fB[j]; b.u[8 + j] = fB[16 + j]; }

        c0 = __builtin_amdgcn_wmma_f32_16x16x32_bf16(false, a0.v, false, b.v, (short)0, c0, false, false);
        c1 = __builtin_amdgcn_wmma_f32_16x16x32_bf16(false, a1.v, false, b.v, (short)0, c1, false, false);
        c2 = __builtin_amdgcn_wmma_f32_16x16x32_bf16(false, a2.v, false, b.v, (short)0, c2, false, false);
        c3 = __builtin_amdgcn_wmma_f32_16x16x32_bf16(false, a3.v, false, b.v, (short)0, c3, false, false);
    };

    for (int step = 0; step < NSTEP; step += 2) {
        body(step, 0);
        body(step + 1, 1);
    }

    // ---- epilogue: C VGPR i -> (M = i + 8*(lane>=16), N = lane&15) ----
    const int n_g  = nbase + wv * 16 + (lane & 15);
    const int moff = (lane >> 4) * 8;
#pragma unroll
    for (int i = 0; i < 8; ++i) {
        const int m0 = 0  + moff + i;
        const int m1 = 16 + moff + i;
        const int m2 = 32 + moff + i;
        const int m3 = 48 + moff + i;
        if (rbase + m0 < cnt) unsafeAtomicAdd(&out[(size_t)stok[m0] * D_MODEL + n_g], sw[m0] * c0[i]);
        if (rbase + m1 < cnt) unsafeAtomicAdd(&out[(size_t)stok[m1] * D_MODEL + n_g], sw[m1] * c1[i]);
        if (rbase + m2 < cnt) unsafeAtomicAdd(&out[(size_t)stok[m2] * D_MODEL + n_g], sw[m2] * c2[i]);
        if (rbase + m3 < cnt) unsafeAtomicAdd(&out[(size_t)stok[m3] * D_MODEL + n_g], sw[m3] * c3[i]);
    }
}

// ---------------- host-side launch ----------------
extern "C" void kernel_launch(void* const* d_in, const int* in_sizes, int n_in,
                              void* d_out, int out_size, void* d_ws, size_t ws_size,
                              hipStream_t stream) {
    (void)in_sizes; (void)n_in; (void)out_size; (void)ws_size;
    const float* x      = (const float*)d_in[0];   // [4,2048,1024]
    const float* gate_w = (const float*)d_in[1];   // [8,1024]
    const float* weight = (const float*)d_in[2];   // [8,1024,1024]
    float*       out    = (float*)d_out;           // [4,2048,1024]

    char* ws = (char*)d_ws;
    size_t o = 0;
    int*   topk_e   = (int*)  (ws + o); o += (size_t)2 * T_TOK * sizeof(int);
    float* topk_w   = (float*)(ws + o); o += (size_t)2 * T_TOK * sizeof(float);
    int*   counts   = (int*)  (ws + o); o += 256;
    int*   tok_list = (int*)  (ws + o); o += (size_t)N_EXP * T_TOK * sizeof(int);
    float* w_list   = (float*)(ws + o); o += (size_t)N_EXP * T_TOK * sizeof(float);

    hipMemsetAsync(out, 0, (size_t)T_TOK * D_MODEL * sizeof(float), stream);

    moe_gate_kernel<<<T_TOK / 8, 256, 0, stream>>>(x, gate_w, topk_e, topk_w);
    moe_compact_kernel<<<N_EXP, 32, 0, stream>>>(topk_e, topk_w, counts, tok_list, w_list);

    dim3 grid(D_MODEL / NT /*8*/, T_TOK / MT /*128*/, N_EXP /*8*/);
    moe_expert_gemm_kernel<<<grid, 256, 0, stream>>>(x, weight, counts, tok_list, w_list, out);
}
